// SchNetStudent_33122787786869
// MI455X (gfx1250) — matrix operand order repaired and checked by reading, not compiled, not run on hardware
//
#include <hip/hip_runtime.h>
#include <hip/hip_bf16.h>

// ---------------------------------------------------------------------------
// SchNet forward for MI455X (gfx1250, wave32, WMMA 16x16x32 f16 + TDM)
// ---------------------------------------------------------------------------

typedef _Float16 half16 __attribute__((ext_vector_type(16)));
typedef _Float16 half8  __attribute__((ext_vector_type(8)));
typedef float    f32x8  __attribute__((ext_vector_type(8)));
typedef unsigned int uint4v __attribute__((ext_vector_type(4)));
typedef int      int8v  __attribute__((ext_vector_type(8)));
typedef int      int4v  __attribute__((ext_vector_type(4)));

#define HIDDIM 64
#define FILTD  128
#define NGAUSS 50
#define LOG2C  0.69314718056f

__device__ __forceinline__ f32x8 wmma16(half16 a, half16 b, f32x8 c) {
  return __builtin_amdgcn_wmma_f32_16x16x32_f16(false, a, false, b, (short)0, c,
                                                false, false);
}

// ShiftedSoftplus: softplus(x) - log(2), numerically stable
__device__ __forceinline__ float sspf(float v) {
  float sp = fmaxf(v, 0.0f) + __logf(1.0f + __expf(-fabsf(v)));
  return sp - LOG2C;
}

// ---------------------------------------------------------------------------
// TDM: 1-D tile DMA of nelem f16 elements global -> LDS (byte offset lds_off).
// Descriptor per CDNA5 ISA D# (CDNA5 ISA):
//   group0: [1:0]=count=1, [63:32]=lds_addr, [120:64]=global_addr, [127:126]=2
//   group1: data_size=1 (2B); tensor_dim0=stride0=nelem; tensor_dim1=tile_dim1=1
// ---------------------------------------------------------------------------
__device__ __forceinline__ void tdm_load_1d_f16(const _Float16* gsrc,
                                                unsigned int lds_off,
                                                int nelem) {
  unsigned long long ga = (unsigned long long)(uintptr_t)gsrc;
  uint4v g0;
  g0.x = 1u;                                     // count=1 (valid user D#)
  g0.y = lds_off;                                // lds_addr (bytes)
  g0.z = (unsigned int)ga;                       // global_addr[31:0]
  g0.w = (unsigned int)(ga >> 32) | 0x80000000u; // global_addr[56:32] | type=2
  int8v g1;
  g1[0] = 0x00010000;                 // workgroup_mask=0, data_size=1 (2 bytes)
  g1[1] = (nelem & 0xFFFF) << 16;     // [63:48] tensor_dim0 lo
  g1[2] = ((nelem >> 16) & 0xFFFF) | (1 << 16);  // td0 hi | tensor_dim1=1
  g1[3] = (nelem & 0xFFFF) << 16;     // [127:112] tile_dim0
  g1[4] = 1;                          // [143:128] tile_dim1 = 1
  g1[5] = nelem;                      // tensor_dim0_stride lo
  g1[6] = 0;
  g1[7] = 0;
  int4v z4 = {0, 0, 0, 0};
#if __clang_major__ >= 23
  int8v z8 = {0, 0, 0, 0, 0, 0, 0, 0};
  __builtin_amdgcn_tensor_load_to_lds(g0, g1, z4, z4, z8, 0);
#else
  __builtin_amdgcn_tensor_load_to_lds(g0, g1, z4, z4, 0);
#endif
}

// B fragment: logical B is [K x 16] stored K-contiguous per output column:
// wt[n * ldk + k].  ISA layout: halves 0-7: K = k0 + b .. +7, halves 8-15:
// K = k0 + b + 16.., b = (lane>=16) ? 8 : 0, N = lane & 15.  -> 2 16-byte loads per lane.
__device__ __forceinline__ half16 load_b_kmaj(const _Float16* __restrict__ wt,
                                              int n, int k0, int lane, int ldk) {
  int kb = k0 + ((lane >= 16) ? 8 : 0);
  const _Float16* p = wt + (size_t)n * ldk + kb;
  union { half16 v; half8 h[2]; } r;
  r.h[0] = *(const half8*)(p);
  r.h[1] = *(const half8*)(p + 16);
  return r.v;
}

// A fragment from row-major [16 x ld] f16 tile (LDS stage).  ISA layout:
// lane m = lane&15; halves 0-7: K = k0 + b .. +7, halves 8-15: K = k0 + b + 16..
// where b = (lane>=16) ? 8 : 0.
__device__ __forceinline__ half16 load_a_rowmaj(const _Float16* s, int m,
                                                int k0, int lane, int ld) {
  int kb = k0 + ((lane >= 16) ? 8 : 0);
  const _Float16* p = s + m * ld + kb;
  union { half16 v; half8 h[2]; } r;
  r.h[0] = *(const half8*)(p);
  r.h[1] = *(const half8*)(p + 16);
  return r.v;
}

// Build an A fragment from a row-major f32 row pointer (global), converting.
__device__ __forceinline__ half16 load_a_f32row(const float* __restrict__ row,
                                                int k0, int lane) {
  int kb = k0 + ((lane >= 16) ? 8 : 0);
  union { half16 v; _Float16 e[16]; } r;
#pragma unroll
  for (int j = 0; j < 8; ++j) {
    r.e[j]     = (_Float16)row[kb + j];
    r.e[j + 8] = (_Float16)row[kb + 16 + j];
  }
  return r.v;
}

// ---------------------------------------------------------------------------
// Utility kernels
// ---------------------------------------------------------------------------
__global__ void zero_f32_kernel(float* __restrict__ p, int n) {
  int i = blockIdx.x * blockDim.x + threadIdx.x;
  if (i < n) p[i] = 0.0f;
}

__global__ void h_init_kernel(const int* __restrict__ z,
                              const float* __restrict__ emb,
                              float* __restrict__ h, int total) {
  int i = blockIdx.x * blockDim.x + threadIdx.x;
  if (i >= total) return;
  int node = i >> 6;  // /64
  int k = i & 63;
  h[i] = emb[z[node] * HIDDIM + k];
}

// Transpose-convert weights: src is [K x Nn] f32 row-major (reference layout),
// dst is [Nn x Kpad] f16 with K contiguous (zero padded K -> Kpad).
__global__ void convert_t_kernel(const float* __restrict__ src,
                                 _Float16* __restrict__ dst,
                                 int K, int Nn, int Kpad) {
  int i = blockIdx.x * blockDim.x + threadIdx.x;
  int total = Nn * Kpad;
  if (i >= total) return;
  int n = i / Kpad, k = i - n * Kpad;
  dst[i] = (k < K) ? (_Float16)src[(size_t)k * Nn + n] : (_Float16)0.0f;
}

// ---------------------------------------------------------------------------
// x = h @ lin1   ([N x 64] @ [64 x 128]) ; one wave per 16-node tile
// ---------------------------------------------------------------------------
__global__ __launch_bounds__(256) void node_lin1_kernel(
    const float* __restrict__ h, const _Float16* __restrict__ lin1t,
    float* __restrict__ x, int nNodes) {
  int lane = threadIdx.x & 31;
  int wave = (blockIdx.x * blockDim.x + threadIdx.x) >> 5;
  int n0 = wave * 16;
  if (n0 >= nNodes) return;
  int m = lane & 15, hi = lane >> 4;

  const float* hrow = &h[(size_t)(n0 + m) * HIDDIM];
  half16 A0 = load_a_f32row(hrow, 0, lane);
  half16 A1 = load_a_f32row(hrow, 32, lane);

#pragma unroll
  for (int nt = 0; nt < 8; ++nt) {
    int ncol = nt * 16 + m;
    f32x8 acc = {};
    acc = wmma16(A0, load_b_kmaj(lin1t, ncol, 0, lane, HIDDIM), acc);
    acc = wmma16(A1, load_b_kmaj(lin1t, ncol, 32, lane, HIDDIM), acc);
#pragma unroll
    for (int r = 0; r < 8; ++r) {
      int M = r + hi * 8;
      x[(size_t)(n0 + M) * FILTD + ncol] = acc[r];
    }
  }
}

// ---------------------------------------------------------------------------
// Edge kernel: weights DMA'd to LDS by TDM once per block; then per 16-edge
// wave: RBF -> filter MLP (2 WMMA GEMMs, B from LDS) -> *cutoff -> x[row]*W ->
// atomic scatter into agg[col].  8 waves/block.
// LDS: 16K (w1t) + 32K (w2t) + 32K (stage) + ~1.5K (edge meta) ~ 83.5 KB
// (well under the 320 KB/workgroup CDNA5 limit; ~3-4 blocks per WGP).
// ---------------------------------------------------------------------------
#define EDGE_WAVES 8
#define SMEM_W1_OFF 0
#define SMEM_W2_OFF (128 * 64 * 2)                      // 16384
#define SMEM_STAGE_OFF (SMEM_W2_OFF + 128 * 128 * 2)    // 49152
#define SMEM_META_OFF (SMEM_STAGE_OFF + EDGE_WAVES * 16 * FILTD * 2)  // 81920
#define SMEM_BYTES (SMEM_META_OFF + EDGE_WAVES * 16 * 12)

__global__ __launch_bounds__(256) void edge_kernel(
    const float* __restrict__ pos, const int* __restrict__ row_idx,
    const int* __restrict__ col_idx, const float* __restrict__ x,
    float* __restrict__ agg, const _Float16* __restrict__ w1t,
    const _Float16* __restrict__ w2t, const float* __restrict__ b1,
    const float* __restrict__ b2, int nEdges) {
  __shared__ __align__(16) unsigned char smem[SMEM_BYTES];
  _Float16* sW1 = (_Float16*)(smem + SMEM_W1_OFF);     // [128][64]
  _Float16* sW2 = (_Float16*)(smem + SMEM_W2_OFF);     // [128][128]
  _Float16* stage = (_Float16*)(smem + SMEM_STAGE_OFF);
  int* srow = (int*)(smem + SMEM_META_OFF);
  int* scol = srow + EDGE_WAVES * 16;
  float* sC = (float*)(scol + EDGE_WAVES * 16);

  // --- Tensor Data Mover: stage filter weights into LDS once per block ---
  if (threadIdx.x == 0) {
    tdm_load_1d_f16(w1t, SMEM_W1_OFF, 128 * 64);
    tdm_load_1d_f16(w2t, SMEM_W2_OFF, 128 * 128);
    __builtin_amdgcn_s_wait_tensorcnt(0);
  }
  __syncthreads();

  const int lane = threadIdx.x & 31;
  const int wib = threadIdx.x >> 5;
  const int wave = blockIdx.x * EDGE_WAVES + wib;
  const int e0 = wave * 16;
  if (e0 >= nEdges) return;

  const int m = lane & 15;
  const int hi = lane >> 4;

  float d = 0.0f;
  if (lane < 16) {
    int e = e0 + lane;
    int rr = row_idx[e];
    int cc = col_idx[e];
    float dx = pos[3 * rr + 0] - pos[3 * cc + 0];
    float dy = pos[3 * rr + 1] - pos[3 * cc + 1];
    float dz = pos[3 * rr + 2] - pos[3 * cc + 2];
    d = __fsqrt_rn(dx * dx + dy * dy + dz * dz);
    srow[wib * 16 + lane] = rr;
    scol[wib * 16 + lane] = cc;
    sC[wib * 16 + lane] = 0.5f * (__cosf(d * 0.31415926535f) + 1.0f);
    __builtin_prefetch(&x[(size_t)rr * FILTD], 0, 0);  // global_prefetch_b8
  }
  __builtin_amdgcn_wave_barrier();

  float dm = __shfl(d, m, 32);

  // RBF A fragments (K = 50 padded to 64), built directly in ISA A-layout.
  const float rstep = 10.0f / 49.0f;
  const float coeff = -0.5f / (rstep * rstep);
  union { half16 v; _Float16 e[16]; } A0, A1;
#pragma unroll
  for (int i = 0; i < 16; ++i) {
    int k = (i & 7) + (hi ? 8 : 0) + ((i >= 8) ? 16 : 0);  // 0..31
    float t0 = dm - (float)k * rstep;
    A0.e[i] = (_Float16)__expf(coeff * t0 * t0);
    int k1 = k + 32;  // 32..63
    if (k1 < NGAUSS) {
      float t1 = dm - (float)k1 * rstep;
      A1.e[i] = (_Float16)__expf(coeff * t1 * t1);
    } else {
      A1.e[i] = (_Float16)0.0f;
    }
  }

  _Float16* st = &stage[wib * 16 * FILTD];

  // GEMM1: [16x64] @ [64x128] (B from LDS), + b1, ssp -> f16 stage
#pragma unroll
  for (int nt = 0; nt < 8; ++nt) {
    int ncol = nt * 16 + m;
    f32x8 acc = {};
    acc = wmma16(A0.v, load_b_kmaj(sW1, ncol, 0, lane, 64), acc);
    acc = wmma16(A1.v, load_b_kmaj(sW1, ncol, 32, lane, 64), acc);
    float bb = b1[ncol];
#pragma unroll
    for (int r = 0; r < 8; ++r) {
      int M = r + hi * 8;
      st[M * FILTD + ncol] = (_Float16)sspf(acc[r] + bb);
    }
  }
  __builtin_amdgcn_wave_barrier();  // LDS is in-order within a wave

  half16 A2[4];
#pragma unroll
  for (int kt = 0; kt < 4; ++kt)
    A2[kt] = load_a_rowmaj(st, m, kt * 32, lane, FILTD);

  // GEMM2: [16x128] @ [128x128] (B from LDS), + b2, * C, gather x[row],
  // scatter into agg[col]
#pragma unroll
  for (int nt = 0; nt < 8; ++nt) {
    int ncol = nt * 16 + m;
    f32x8 acc = {};
#pragma unroll
    for (int kt = 0; kt < 4; ++kt)
      acc = wmma16(A2[kt], load_b_kmaj(sW2, ncol, kt * 32, lane, FILTD), acc);
    float bb = b2[ncol];
#pragma unroll
    for (int r = 0; r < 8; ++r) {
      int M = r + hi * 8;
      float W = (acc[r] + bb) * sC[wib * 16 + M];
      int rn = srow[wib * 16 + M];
      int cn = scol[wib * 16 + M];
      float xv = x[(size_t)rn * FILTD + ncol];
      atomicAdd(&agg[(size_t)cn * FILTD + ncol], xv * W);
    }
  }
}

// ---------------------------------------------------------------------------
// Node update: x2 = agg@lin2 + b2 ; h += ssp(x2)@lin + b.  8 waves/block.
// ---------------------------------------------------------------------------
__global__ __launch_bounds__(256) void node_update_kernel(
    const float* __restrict__ agg, const _Float16* __restrict__ lin2t,
    const float* __restrict__ lin2_b, const _Float16* __restrict__ lintw,
    const float* __restrict__ lin_b, float* __restrict__ h, int nNodes) {
  __shared__ _Float16 stage[8 * 16 * HIDDIM];  // 16 KB
  int lane = threadIdx.x & 31;
  int wib = threadIdx.x >> 5;
  int wave = blockIdx.x * 8 + wib;
  int n0 = wave * 16;
  if (n0 >= nNodes) return;
  int m = lane & 15, hi = lane >> 4;

  const float* arow = &agg[(size_t)(n0 + m) * FILTD];
  half16 A[4];
#pragma unroll
  for (int kt = 0; kt < 4; ++kt) A[kt] = load_a_f32row(arow, kt * 32, lane);

  _Float16* st = &stage[wib * 16 * HIDDIM];
#pragma unroll
  for (int nt = 0; nt < 4; ++nt) {  // HID = 64 -> 4 N-tiles
    int ncol = nt * 16 + m;
    f32x8 acc = {};
#pragma unroll
    for (int kt = 0; kt < 4; ++kt)
      acc = wmma16(A[kt], load_b_kmaj(lin2t, ncol, kt * 32, lane, FILTD), acc);
    float bb = lin2_b[ncol];
#pragma unroll
    for (int r = 0; r < 8; ++r) {
      int M = r + hi * 8;
      st[M * HIDDIM + ncol] = (_Float16)sspf(acc[r] + bb);
    }
  }
  __builtin_amdgcn_wave_barrier();

  half16 A2[2];
#pragma unroll
  for (int kt = 0; kt < 2; ++kt)
    A2[kt] = load_a_rowmaj(st, m, kt * 32, lane, HIDDIM);

#pragma unroll
  for (int nt = 0; nt < 4; ++nt) {
    int ncol = nt * 16 + m;
    f32x8 acc = {};
    acc = wmma16(A2[0], load_b_kmaj(lintw, ncol, 0, lane, HIDDIM), acc);
    acc = wmma16(A2[1], load_b_kmaj(lintw, ncol, 32, lane, HIDDIM), acc);
    float bb = lin_b[ncol];
#pragma unroll
    for (int r = 0; r < 8; ++r) {
      int M = r + hi * 8;
      h[(size_t)(n0 + M) * HIDDIM + ncol] += acc[r] + bb;  // residual
    }
  }
}

// ---------------------------------------------------------------------------
// Output head: per node ssp(h@out_w1+b1)@out_w2+b2 -> atomicAdd(out[batch])
// ---------------------------------------------------------------------------
__global__ __launch_bounds__(256) void output_kernel(
    const float* __restrict__ h, const _Float16* __restrict__ out1t,
    const float* __restrict__ out_b1, const float* __restrict__ out_w2,
    const float* __restrict__ out_b2, const int* __restrict__ batch,
    float* __restrict__ out, int nNodes) {
  int lane = threadIdx.x & 31;
  int wave = (blockIdx.x * blockDim.x + threadIdx.x) >> 5;
  int n0 = wave * 16;
  if (n0 >= nNodes) return;
  int m = lane & 15, hi = lane >> 4;

  const float* hrow = &h[(size_t)(n0 + m) * HIDDIM];
  half16 A0 = load_a_f32row(hrow, 0, lane);
  half16 A1 = load_a_f32row(hrow, 32, lane);

  float p[8];
#pragma unroll
  for (int r = 0; r < 8; ++r) p[r] = 0.0f;

#pragma unroll
  for (int nt = 0; nt < 2; ++nt) {  // 32 cols
    int ncol = nt * 16 + m;
    f32x8 acc = {};
    acc = wmma16(A0, load_b_kmaj(out1t, ncol, 0, lane, HIDDIM), acc);
    acc = wmma16(A1, load_b_kmaj(out1t, ncol, 32, lane, HIDDIM), acc);
    float bb = out_b1[ncol];
    float w2v = out_w2[ncol];
#pragma unroll
    for (int r = 0; r < 8; ++r) p[r] += sspf(acc[r] + bb) * w2v;
  }

  // Reduce across the 16 lanes of each half-wave (C-layout: N = lane&15)
  float b2v = out_b2[0];
#pragma unroll
  for (int r = 0; r < 8; ++r) {
    float v = p[r];
#pragma unroll
    for (int o = 1; o < 16; o <<= 1) v += __shfl_xor(v, o, 32);
    if (m == 0) {
      int node = n0 + r + hi * 8;
      atomicAdd(&out[batch[node]], v + b2v);
    }
  }
}

// ---------------------------------------------------------------------------
// Host launcher
// ---------------------------------------------------------------------------
extern "C" void kernel_launch(void* const* d_in, const int* in_sizes, int n_in,
                              void* d_out, int out_size, void* d_ws,
                              size_t ws_size, hipStream_t stream) {
  (void)n_in; (void)ws_size;
  const int*   z      = (const int*)  d_in[0];
  const float* pos    = (const float*)d_in[1];
  const int*   eidx   = (const int*)  d_in[2];
  const int*   batch  = (const int*)  d_in[3];
  const float* emb    = (const float*)d_in[4];
  const float* mlp_w1 = (const float*)d_in[5];
  const float* mlp_b1 = (const float*)d_in[6];
  const float* mlp_w2 = (const float*)d_in[7];
  const float* mlp_b2 = (const float*)d_in[8];
  const float* lin1_w = (const float*)d_in[9];
  const float* lin2_w = (const float*)d_in[10];
  const float* lin2_b = (const float*)d_in[11];
  const float* lin_w  = (const float*)d_in[12];
  const float* lin_b  = (const float*)d_in[13];
  const float* out_w1 = (const float*)d_in[14];
  const float* out_b1 = (const float*)d_in[15];
  const float* out_w2 = (const float*)d_in[16];
  const float* out_b2 = (const float*)d_in[17];
  float* out = (float*)d_out;

  const int nN = in_sizes[0];
  const int nE = in_sizes[2] / 2;
  const int nB = out_size;
  const int* row = eidx;
  const int* col = eidx + nE;

  // workspace layout
  float* h   = (float*)d_ws;                         // N x 64
  float* x   = h + (size_t)nN * HIDDIM;              // N x 128
  float* agg = x + (size_t)nN * FILTD;               // N x 128
  _Float16* lin1t = (_Float16*)(agg + (size_t)nN * FILTD);  // 2 x [128][64]
  _Float16* w1t   = lin1t + 2 * 128 * 64;            // 2 x [128][64]
  _Float16* w2t   = w1t + 2 * 128 * 64;              // 2 x [128][128]
  _Float16* lin2t = w2t + 2 * 128 * 128;             // 2 x [64][128]
  _Float16* lintw = lin2t + 2 * 64 * 128;            // 2 x [64][64]
  _Float16* out1t = lintw + 2 * 64 * 64;             // [32][64]

  auto cgrid = [](int total) { return (total + 255) / 256; };

  // weight conversion (f32 [K][N] -> f16 [N][Kpad], K-contiguous)
  for (int l = 0; l < 2; ++l) {
    convert_t_kernel<<<cgrid(128 * 64), 256, 0, stream>>>(
        lin1_w + (size_t)l * HIDDIM * FILTD, lin1t + (size_t)l * 128 * 64,
        64, 128, 64);
    convert_t_kernel<<<cgrid(128 * 64), 256, 0, stream>>>(
        mlp_w1 + (size_t)l * NGAUSS * FILTD, w1t + (size_t)l * 128 * 64,
        50, 128, 64);
    convert_t_kernel<<<cgrid(128 * 128), 256, 0, stream>>>(
        mlp_w2 + (size_t)l * FILTD * FILTD, w2t + (size_t)l * 128 * 128,
        128, 128, 128);
    convert_t_kernel<<<cgrid(64 * 128), 256, 0, stream>>>(
        lin2_w + (size_t)l * FILTD * HIDDIM, lin2t + (size_t)l * 64 * 128,
        128, 64, 128);
    convert_t_kernel<<<cgrid(64 * 64), 256, 0, stream>>>(
        lin_w + (size_t)l * HIDDIM * HIDDIM, lintw + (size_t)l * 64 * 64,
        64, 64, 64);
  }
  convert_t_kernel<<<cgrid(32 * 64), 256, 0, stream>>>(out_w1, out1t, 64, 32, 64);

  h_init_kernel<<<cgrid(nN * HIDDIM), 256, 0, stream>>>(z, emb, h, nN * HIDDIM);
  zero_f32_kernel<<<cgrid(nB), 256, 0, stream>>>(out, nB);

  const int nodeWaves = (nN + 15) / 16;
  const int nodeBlocks = (nodeWaves + 7) / 8;
  const int edgeWaves = (nE + 15) / 16;
  const int edgeBlocks = (edgeWaves + EDGE_WAVES - 1) / EDGE_WAVES;

  for (int l = 0; l < 2; ++l) {
    zero_f32_kernel<<<cgrid(nN * FILTD), 256, 0, stream>>>(agg, nN * FILTD);
    node_lin1_kernel<<<nodeBlocks, 256, 0, stream>>>(
        h, lin1t + (size_t)l * 128 * 64, x, nN);
    edge_kernel<<<edgeBlocks, 256, 0, stream>>>(
        pos, row, col, x, agg, w1t + (size_t)l * 128 * 64,
        w2t + (size_t)l * 128 * 128, mlp_b1 + (size_t)l * FILTD,
        mlp_b2 + (size_t)l * FILTD, nE);
    node_update_kernel<<<nodeBlocks, 256, 0, stream>>>(
        agg, lin2t + (size_t)l * 64 * 128, lin2_b + (size_t)l * HIDDIM,
        lintw + (size_t)l * 64 * 64, lin_b + (size_t)l * HIDDIM, h, nN);
  }

  output_kernel<<<nodeBlocks, 256, 0, stream>>>(h, out1t, out_b1, out_w2,
                                                out_b2, batch, out, nN);
}
